// LumenAttention_22969485099758
// MI455X (gfx1250) — matrix operand order repaired and checked by reading, not compiled, not run on hardware
//
#include <hip/hip_runtime.h>
#include <hip/hip_bf16.h>
#include <math.h>

typedef _Float16 v16h __attribute__((ext_vector_type(16)));
typedef _Float16 h8   __attribute__((ext_vector_type(8)));
typedef float    v8f  __attribute__((ext_vector_type(8)));

#define WL   512        // sliding window (past keys)
#define SEQ  2048
#define NH   16
#define HD   128
#define QT   128        // queries per workgroup (8 waves x 16)
#define KC   32         // keys per chunk (matches WMMA K=32 for PV)

__global__ __launch_bounds__(256)
void lumen_attn_swa_alibi(const float* __restrict__ Qg,
                          const float* __restrict__ Kg,
                          const float* __restrict__ Vg,
                          const float* __restrict__ slopes,
                          float* __restrict__ Og)
{
    // K chunk row-major (keys x dims), V chunk transposed (dims x keys),
    // per-wave P bounce buffer for C-layout -> A-layout conversion.
    __shared__ _Float16 Kl[KC][136];      // 272B row stride, 16B aligned
    __shared__ _Float16 Vt[HD][40];       // 80B row stride, 16B aligned
    __shared__ _Float16 Pl[8][16][40];

    const int tid  = threadIdx.x;
    const int wave = tid >> 5;
    const int lane = tid & 31;
    const int hi   = lane >> 4;       // which 16-lane half-group
    const int colN = lane & 15;

    const int blk   = blockIdx.x;
    const int qtile = blk % (SEQ / QT);
    const int h     = (blk / (SEQ / QT)) % NH;
    const int b     = blk / ((SEQ / QT) * NH);
    const int q0    = qtile * QT;
    const int qw    = q0 + wave * 16;     // this wave's first query row

    const float scale = 0.08838834764831845f;   // 1/sqrt(128)
    const float slope = slopes[h];

    // ---- Q fragments (A-layout, 16x32 f16 per 32-dim slab), scale folded in
    v16h qf[4];
    {
        const int qrow = qw + colN;       // A-layout: row M = lane%16
        const float* qp = Qg + (((size_t)(b * SEQ + qrow) * NH + h) * HD);
        #pragma unroll
        for (int f = 0; f < 4; ++f) {
            const int base = 32 * f + (hi ? 16 : 0);
            union { v16h v; _Float16 e[16]; } u;
            #pragma unroll
            for (int j = 0; j < 8; ++j) {
                u.e[j]     = (_Float16)(qp[base + j]      * scale);
                u.e[j + 8] = (_Float16)(qp[base + 8 + j] * scale);
            }
            qf[f] = u.v;
        }
    }

    // ---- online-softmax state (row stats replicated across 16-lane group)
    float m[8], l[8];
    v8f o[8];
    #pragma unroll
    for (int i = 0; i < 8; ++i) { m[i] = -1e30f; l[i] = 0.0f; }
    #pragma unroll
    for (int n = 0; n < 8; ++n) {
        #pragma unroll
        for (int e = 0; e < 8; ++e) o[n][e] = 0.0f;
    }

    int kstart = q0 - WL; if (kstart < 0) kstart = 0;   // multiple of 32
    const int kend = q0 + QT;                           // exclusive

    for (int kc = kstart; kc < kend; kc += KC) {
        __syncthreads();   // previous chunk's compute done before overwrite

        // ---- cooperative load: 32 keys x 128 dims of K and V, f32 -> f16
        {
            const int key = tid >> 3;            // 0..31
            const int dB  = (tid & 7) * 16;      // 0,16,...,112
            const size_t gb = ((size_t)(b * SEQ + kc + key) * NH + h) * HD + dB;
            const float4* kp4 = (const float4*)(Kg + gb);
            const float4* vp4 = (const float4*)(Vg + gb);
            float kv[16], vv[16];
            #pragma unroll
            for (int t = 0; t < 4; ++t) {
                float4 kk = kp4[t], vx = vp4[t];
                kv[4*t+0]=kk.x; kv[4*t+1]=kk.y; kv[4*t+2]=kk.z; kv[4*t+3]=kk.w;
                vv[4*t+0]=vx.x; vv[4*t+1]=vx.y; vv[4*t+2]=vx.z; vv[4*t+3]=vx.w;
            }
            union { h8 v; _Float16 e[8]; } lo, hi2;
            #pragma unroll
            for (int j = 0; j < 8; ++j) { lo.e[j]=(_Float16)kv[j]; hi2.e[j]=(_Float16)kv[8+j]; }
            *(h8*)&Kl[key][dB]     = lo.v;
            *(h8*)&Kl[key][dB + 8] = hi2.v;
            #pragma unroll
            for (int j = 0; j < 16; ++j) Vt[dB + j][key] = (_Float16)vv[j];
        }
        __syncthreads();

        // wave-uniform window check -> EXEC stays all-ones inside
        if (kc <= qw + 15 && kc + KC - 1 >= qw - WL) {
            // ---- S = Q K^T for two 16-key tiles (K-dim = 128 in 4 slabs)
            v8f s0, s1;
            #pragma unroll
            for (int e = 0; e < 8; ++e) { s0[e] = 0.0f; s1[e] = 0.0f; }
            #pragma unroll
            for (int f = 0; f < 4; ++f) {
                const int d0 = 32 * f + (hi ? 16 : 0);   // B-layout contraction slots
                union { v16h v; h8 p[2]; } b0, b1;
                b0.p[0] = *(const h8*)&Kl[colN][d0];
                b0.p[1] = *(const h8*)&Kl[colN][d0 + 8];
                b1.p[0] = *(const h8*)&Kl[colN + 16][d0];
                b1.p[1] = *(const h8*)&Kl[colN + 16][d0 + 8];
                s0 = __builtin_amdgcn_wmma_f32_16x16x32_f16(false, qf[f], false, b0.v,
                                                            (short)0, s0, false, false);
                s1 = __builtin_amdgcn_wmma_f32_16x16x32_f16(false, qf[f], false, b1.v,
                                                            (short)0, s1, false, false);
            }

            // ALiBi: slope*(k-q); the -slope*q part is constant per softmax row
            // and cancels exactly -> only slope*k matters, one scalar per lane.
            const float bias0 = slope * (float)(kc + colN);
            const float bias1 = bias0 + slope * 16.0f;

            float p0[8], p1[8], alpha[8];

            // chunk fully inside window for ALL 16 rows of this wave?
            const bool interior = (kc + KC - 1 <= qw) && (kc >= qw + 15 - WL);

            if (interior) {
                // ---- fast path: no masks, just bias + online softmax
                #pragma unroll
                for (int i = 0; i < 8; ++i) {
                    const float sc0 = s0[i] + bias0;
                    const float sc1 = s1[i] + bias1;
                    float cm = fmaxf(sc0, sc1);
                    cm = fmaxf(cm, __shfl_xor(cm, 1, 32));
                    cm = fmaxf(cm, __shfl_xor(cm, 2, 32));
                    cm = fmaxf(cm, __shfl_xor(cm, 4, 32));
                    cm = fmaxf(cm, __shfl_xor(cm, 8, 32));
                    const float mn = fmaxf(m[i], cm);
                    alpha[i] = __expf(m[i] - mn);
                    m[i] = mn;
                    p0[i] = __expf(sc0 - mn);
                    p1[i] = __expf(sc1 - mn);
                    float rs = p0[i] + p1[i];
                    rs += __shfl_xor(rs, 1, 32);
                    rs += __shfl_xor(rs, 2, 32);
                    rs += __shfl_xor(rs, 4, 32);
                    rs += __shfl_xor(rs, 8, 32);
                    l[i] = l[i] * alpha[i] + rs;
                }
            } else {
                // ---- boundary path: per-element causal + window masks
                #pragma unroll
                for (int i = 0; i < 8; ++i) {
                    const int qv = qw + i + (hi ? 8 : 0);
                    const int r0 = kc + colN - qv;          // <=0 && >=-WL allowed
                    const int r1 = r0 + 16;
                    const bool ok0 = (r0 <= 0) && (r0 >= -WL);
                    const bool ok1 = (r1 <= 0) && (r1 >= -WL);
                    const float sc0 = ok0 ? (s0[i] + bias0) : -1e30f;
                    const float sc1 = ok1 ? (s1[i] + bias1) : -1e30f;
                    float cm = fmaxf(sc0, sc1);
                    cm = fmaxf(cm, __shfl_xor(cm, 1, 32));
                    cm = fmaxf(cm, __shfl_xor(cm, 2, 32));
                    cm = fmaxf(cm, __shfl_xor(cm, 4, 32));
                    cm = fmaxf(cm, __shfl_xor(cm, 8, 32));
                    const float mn = fmaxf(m[i], cm);
                    alpha[i] = __expf(m[i] - mn);
                    m[i] = mn;
                    p0[i] = ok0 ? __expf(sc0 - mn) : 0.0f;  // p=0 on masked rows
                    p1[i] = ok1 ? __expf(sc1 - mn) : 0.0f;
                    float rs = p0[i] + p1[i];
                    rs += __shfl_xor(rs, 1, 32);
                    rs += __shfl_xor(rs, 2, 32);
                    rs += __shfl_xor(rs, 4, 32);
                    rs += __shfl_xor(rs, 8, 32);
                    l[i] = l[i] * alpha[i] + rs;
                }
            }

            // ---- P: C-layout -> A-layout via wave-private LDS bounce
            #pragma unroll
            for (int i = 0; i < 8; ++i) {
                const int M = i + (hi ? 8 : 0);
                Pl[wave][M][colN]      = (_Float16)p0[i];
                Pl[wave][M][colN + 16] = (_Float16)p1[i];
            }
            union { v16h v; h8 p[2]; } pa;
            {
                const int base = hi ? 16 : 0;    // A-layout K slots for this half-group
                pa.p[0] = *(const h8*)&Pl[wave][colN][base];
                pa.p[1] = *(const h8*)&Pl[wave][colN][base + 8];
            }

            // ---- O = diag(alpha)*O + P V  (8 dim-tiles, single K=32 WMMA each)
            #pragma unroll
            for (int n = 0; n < 8; ++n) {
                v8f oo = o[n];
                #pragma unroll
                for (int i = 0; i < 8; ++i) oo[i] *= alpha[i];
                union { v16h v; h8 p[2]; } bv;
                const int dim = 16 * n + colN;      // B-layout column
                const int kb  = hi ? 16 : 0;        // contraction (key) slots
                bv.p[0] = *(const h8*)&Vt[dim][kb];
                bv.p[1] = *(const h8*)&Vt[dim][kb + 8];
                o[n] = __builtin_amdgcn_wmma_f32_16x16x32_f16(false, pa.v, false, bv.v,
                                                              (short)0, oo, false, false);
            }
        }
    }

    // ---- epilogue: normalize and store fp32 output [B,S,H,D]
    float invl[8];
    #pragma unroll
    for (int i = 0; i < 8; ++i) invl[i] = 1.0f / l[i];
    #pragma unroll
    for (int n = 0; n < 8; ++n) {
        const int dim = 16 * n + colN;
        #pragma unroll
        for (int i = 0; i < 8; ++i) {
            const int qv = qw + i + (hi ? 8 : 0);
            Og[((size_t)(b * SEQ + qv) * NH + h) * HD + dim] = o[n][i] * invl[i];
        }
    }
}

extern "C" void kernel_launch(void* const* d_in, const int* in_sizes, int n_in,
                              void* d_out, int out_size, void* d_ws, size_t ws_size,
                              hipStream_t stream) {
    const float* q = (const float*)d_in[0];
    const float* k = (const float*)d_in[1];
    const float* v = (const float*)d_in[2];
    const float* s = (const float*)d_in[3];
    float* out = (float*)d_out;

    const int B = in_sizes[0] / (SEQ * NH * HD);
    const int grid = B * NH * (SEQ / QT);
    lumen_attn_swa_alibi<<<grid, 256, 0, stream>>>(q, k, v, s, out);
}
